// BatchTemporalContrastiveLoss_13357348291322
// MI455X (gfx1250) — matrix operand order repaired and checked by reading, not compiled, not run on hardware
//
#include <hip/hip_runtime.h>

typedef __attribute__((ext_vector_type(16))) _Float16 v16h;
typedef __attribute__((ext_vector_type(8)))  float    v8f;

#define HIDDEN 512
#define MID    256
#define PROJ   128
#define NEGK   8
#define TEMP_INV (1.0f / 0.07f)
#define EPS_LN 1e-5f

// LDS tile: 16 rows x 260 floats (stride 260 -> 4-bank row offset, conflict-free)
#define HSTRIDE 260

// ---------------------------------------------------------------------------
// Kernel 0: pre-swizzle a weight matrix W[K,N] (f32, row-major) into hi/lo f16
// arrays laid out exactly as v_wmma_f32_16x16x32_f16 B-fragments:
//   flat idx = (((kb*(N/16) + j)*32 + lane)*16 + e)
//   value    = W[kb*32 + 8*(lane>>4) + (e&7) + 16*(e>>3)][j*16 + (lane&15)]
// so each lane's 16-half fragment is one contiguous 32-byte run (2x b128).
// Split precision: w = hi + lo, hi = f16(w), lo = f16(w - hi).
// ---------------------------------------------------------------------------
__global__ __launch_bounds__(256) void ctl_swizzle_kernel(
    const float* __restrict__ W, _Float16* __restrict__ Whi,
    _Float16* __restrict__ Wlo, int K, int N)
{
    const int idx = blockIdx.x * 256 + threadIdx.x;
    if (idx >= K * N) return;
    const int e    = idx & 15;
    const int lane = (idx >> 4) & 31;
    const int t    = idx >> 9;          // kb*(N/16) + j
    const int nt   = N >> 4;
    const int j    = t % nt;
    const int kb   = t / nt;
    const int k    = kb * 32 + ((lane >> 4) << 3) + (e & 7) + ((e >> 3) << 4);
    const int n    = j * 16 + (lane & 15);
    const float w  = W[(size_t)k * N + n];
    const _Float16 hi = (_Float16)w;
    Whi[idx] = hi;
    Wlo[idx] = (_Float16)(w - (float)hi);
}

// Build an A-fragment pair (hi/lo) for one K=32 block from 16 consecutive f32
// values: runs [base .. base+7] and [base+16 .. base+23] per the ISA A layout
// (lane<16: K0-7 / K16-23; lane>=16: K8-15 / K24-31 via `off`).
__device__ __forceinline__ void make_a_frag(const float* __restrict__ src,
                                            v16h& ah, v16h& al)
{
#pragma unroll
    for (int e = 0; e < 8; ++e) {
        const float x0 = src[e];
        const float x1 = src[16 + e];
        const _Float16 h0 = (_Float16)x0;
        const _Float16 h1 = (_Float16)x1;
        ah[e]     = h0;
        ah[8 + e] = h1;
        al[e]     = (_Float16)(x0 - (float)h0);
        al[8 + e] = (_Float16)(x1 - (float)h1);
    }
}

// ---------------------------------------------------------------------------
// Kernel 1: fused projection  X[16 rows,512] -> relu(XW1+b1) -> (.)W2+b2
//           -> LayerNorm -> L2 normalize -> pool row
// One wave (32 threads) per block; one 16-row tile per wave.
// Split-f16 WMMA: acc += ah*bh + ah*bl + al*bh  (error ~2^-22, ~fp32)
// ---------------------------------------------------------------------------
__global__ __launch_bounds__(32) void ctl_project_kernel(
    const float* __restrict__ X0, const float* __restrict__ X1,
    const _Float16* __restrict__ W1h, const _Float16* __restrict__ W1l,
    const _Float16* __restrict__ W2h, const _Float16* __restrict__ W2l,
    const float* __restrict__ b1, const float* __restrict__ b2,
    const float* __restrict__ g,  const float* __restrict__ beta,
    float* __restrict__ pool, int B)
{
    __shared__ float hlds[16][HSTRIDE];

    const int lane  = threadIdx.x;          // 0..31 (wave32)
    const int mrow  = lane & 15;            // M (A) / N (B,C) sub-index
    const int off   = (lane >> 4) * 8;      // A-frag K sub-offset (0 or 8)
    const int mbase = (lane >> 4) * 8;      // C layout: lanes 16-31 hold M=v+8

    const int nb      = B >> 4;
    const int srcSel  = (blockIdx.x >= (unsigned)nb) ? 1 : 0;
    const int tile    = blockIdx.x - srcSel * nb;
    const int rowBase = tile * 16;
    const float* __restrict__ X = srcSel ? X1 : X0;
    const float* __restrict__ xrow = X + (size_t)(rowBase + mrow) * HIDDEN;

    // -------- GEMM1: H[16x256] = relu(X[16x512] @ W1 + b1), K-blocks of 32 --
    v8f acc[16] = {};
    for (int kb = 0; kb < HIDDEN / 32; ++kb) {
        v16h ah, al;
        make_a_frag(xrow + kb * 32 + off, ah, al);
        // B-frag base for this K-block: 16 tiles of 32 lanes * 16 halves
        const size_t fb = ((size_t)kb * 16 * 32 + lane) * 16;
#pragma unroll
        for (int j = 0; j < 16; ++j) {
            const v16h bh = *(const v16h*)(W1h + fb + (size_t)j * 32 * 16);
            const v16h bl = *(const v16h*)(W1l + fb + (size_t)j * 32 * 16);
            acc[j] = __builtin_amdgcn_wmma_f32_16x16x32_f16(
                false, ah, false, bh, (short)0, acc[j], false, false);
            acc[j] = __builtin_amdgcn_wmma_f32_16x16x32_f16(
                false, ah, false, bl, (short)0, acc[j], false, false);
            acc[j] = __builtin_amdgcn_wmma_f32_16x16x32_f16(
                false, al, false, bh, (short)0, acc[j], false, false);
        }
    }

    // bias + ReLU, write H tile to LDS (C layout -> row-major LDS)
#pragma unroll
    for (int j = 0; j < 16; ++j) {
        const int n = j * 16 + mrow;
        const float bias = b1[n];
#pragma unroll
        for (int v = 0; v < 8; ++v) {
            hlds[mbase + v][n] = fmaxf(acc[j][v] + bias, 0.0f);
        }
    }
    __syncthreads();

    // -------- GEMM2: P[16x128] = H[16x256] @ W2 + b2 --------
    v8f acc2[8] = {};
    for (int kb = 0; kb < MID / 32; ++kb) {
        v16h ah, al;
        make_a_frag(&hlds[mrow][kb * 32 + off], ah, al);
        const size_t fb = ((size_t)kb * 8 * 32 + lane) * 16;
#pragma unroll
        for (int j = 0; j < 8; ++j) {
            const v16h bh = *(const v16h*)(W2h + fb + (size_t)j * 32 * 16);
            const v16h bl = *(const v16h*)(W2l + fb + (size_t)j * 32 * 16);
            acc2[j] = __builtin_amdgcn_wmma_f32_16x16x32_f16(
                false, ah, false, bh, (short)0, acc2[j], false, false);
            acc2[j] = __builtin_amdgcn_wmma_f32_16x16x32_f16(
                false, ah, false, bl, (short)0, acc2[j], false, false);
            acc2[j] = __builtin_amdgcn_wmma_f32_16x16x32_f16(
                false, al, false, bh, (short)0, acc2[j], false, false);
        }
    }
    __syncthreads();

    // bias2, store P tile to LDS (reuse hlds, same stride)
#pragma unroll
    for (int j = 0; j < 8; ++j) {
        const int n = j * 16 + mrow;
        const float bias = b2[n];
#pragma unroll
        for (int v = 0; v < 8; ++v) {
            hlds[mbase + v][n] = acc2[j][v] + bias;
        }
    }
    __syncthreads();

    // -------- LayerNorm + L2 normalize, one row per lane (lanes 0..15) -----
    if (lane < 16) {
        float sum = 0.0f, sq = 0.0f;
        for (int c = 0; c < PROJ; ++c) {
            const float x = hlds[lane][c];
            sum += x;
            sq  += x * x;
        }
        const float mu   = sum * (1.0f / PROJ);
        const float var  = sq * (1.0f / PROJ) - mu * mu;
        const float rstd = rsqrtf(var + EPS_LN);

        float s2 = 0.0f;
        for (int c = 0; c < PROJ; ++c) {
            const float v = (hlds[lane][c] - mu) * rstd * g[c] + beta[c];
            hlds[lane][c] = v;
            s2 += v * v;
        }
        const float inv = rsqrtf(s2);

        float* __restrict__ out = pool + ((size_t)srcSel * B + rowBase + lane) * PROJ;
        for (int c = 0; c < PROJ; ++c) out[c] = hlds[lane][c] * inv;
    }
}

// ---------------------------------------------------------------------------
// Kernel 2: per-row InfoNCE loss. One wave per row; lane l holds float4 slice.
// ---------------------------------------------------------------------------
__global__ __launch_bounds__(256) void ctl_loss_kernel(
    const float* __restrict__ pool, const int* __restrict__ neg_idx,
    float* __restrict__ losses, int B)
{
    const int lane = threadIdx.x & 31;
    const int row  = blockIdx.x * 8 + (threadIdx.x >> 5);
    if (row >= B) return;

    const float4 av = ((const float4*)(pool + (size_t)row * PROJ))[lane];

    float sims[1 + NEGK];
    {
        const float4 pv = ((const float4*)(pool + (size_t)(B + row) * PROJ))[lane];
        float part = av.x * pv.x + av.y * pv.y + av.z * pv.z + av.w * pv.w;
#pragma unroll
        for (int off = 16; off > 0; off >>= 1) part += __shfl_xor(part, off, 32);
        sims[0] = part * TEMP_INV;
    }
#pragma unroll
    for (int k = 0; k < NEGK; ++k) {
        const int idx = neg_idx[(size_t)row * NEGK + k];
        const float4 nv = ((const float4*)(pool + (size_t)idx * PROJ))[lane];
        float part = av.x * nv.x + av.y * nv.y + av.z * nv.z + av.w * nv.w;
#pragma unroll
        for (int off = 16; off > 0; off >>= 1) part += __shfl_xor(part, off, 32);
        sims[k + 1] = part * TEMP_INV;
    }

    float m = sims[0];
#pragma unroll
    for (int i = 1; i <= NEGK; ++i) m = fmaxf(m, sims[i]);
    float s = 0.0f;
#pragma unroll
    for (int i = 0; i <= NEGK; ++i) s += expf(sims[i] - m);
    const float loss = -sims[0] + m + logf(s);

    if (lane == 0) losses[row] = loss;
}

// ---------------------------------------------------------------------------
// Kernel 3: deterministic fixed-order mean reduction (single block).
// ---------------------------------------------------------------------------
__global__ __launch_bounds__(256) void ctl_reduce_kernel(
    const float* __restrict__ losses, float* __restrict__ out, int B)
{
    __shared__ float smem[256];
    const int t = threadIdx.x;
    float s = 0.0f;
    for (int i = t; i < B; i += 256) s += losses[i];
    smem[t] = s;
    __syncthreads();
    for (int off = 128; off > 0; off >>= 1) {
        if (t < off) smem[t] += smem[t + off];
        __syncthreads();
    }
    if (t == 0) out[0] = smem[0] / (float)B;
}

// ---------------------------------------------------------------------------
extern "C" void kernel_launch(void* const* d_in, const int* in_sizes, int n_in,
                              void* d_out, int out_size, void* d_ws, size_t ws_size,
                              hipStream_t stream) {
    const float* hidden_states   = (const float*)d_in[0];
    const float* positive_hidden = (const float*)d_in[1];
    const int*   neg_idx         = (const int*)d_in[2];
    const float* W1   = (const float*)d_in[3];
    const float* b1   = (const float*)d_in[4];
    const float* W2   = (const float*)d_in[5];
    const float* b2   = (const float*)d_in[6];
    const float* g    = (const float*)d_in[7];
    const float* beta = (const float*)d_in[8];
    float* out = (float*)d_out;

    const int B = in_sizes[0] / HIDDEN;   // 65536

    // workspace layout (all offsets 256B-aligned):
    //   pool   [2B,128] f32
    //   losses [B]      f32
    //   W1hi/W1lo [512*256] f16 (swizzled), W2hi/W2lo [256*128] f16 (swizzled)
    float* pool   = (float*)d_ws;
    float* losses = pool + (size_t)2 * B * PROJ;
    _Float16* W1h = (_Float16*)(losses + B);
    _Float16* W1l = W1h + (size_t)HIDDEN * MID;
    _Float16* W2h = W1l + (size_t)HIDDEN * MID;
    _Float16* W2l = W2h + (size_t)MID * PROJ;

    ctl_swizzle_kernel<<<dim3((HIDDEN * MID + 255) / 256), dim3(256), 0, stream>>>(
        W1, W1h, W1l, HIDDEN, MID);
    ctl_swizzle_kernel<<<dim3((MID * PROJ + 255) / 256), dim3(256), 0, stream>>>(
        W2, W2h, W2l, MID, PROJ);

    const int nb = B / 16;                 // 16-row tiles per source
    ctl_project_kernel<<<dim3(2 * nb), dim3(32), 0, stream>>>(
        hidden_states, positive_hidden, W1h, W1l, W2h, W2l,
        b1, b2, g, beta, pool, B);

    ctl_loss_kernel<<<dim3((B + 7) / 8), dim3(256), 0, stream>>>(
        pool, neg_idx, losses, B);

    ctl_reduce_kernel<<<dim3(1), dim3(256), 0, stream>>>(losses, out, B);
}
